// Conv2DMultiHeadAttention_88983132438701
// MI455X (gfx1250) — matrix operand order repaired and checked by reading, not compiled, not run on hardware
//
#include <hip/hip_runtime.h>
#include <hip/hip_bf16.h>
#include <math.h>

typedef __bf16 bf16;
typedef __attribute__((ext_vector_type(16))) __bf16 v16bf;
typedef __attribute__((ext_vector_type(8)))  __bf16 v8bf;
typedef __attribute__((ext_vector_type(8)))  float  v8f;
typedef __attribute__((ext_vector_type(4)))  unsigned int v4u;
typedef __attribute__((ext_vector_type(8)))  unsigned int v8u;

#define C_DIM 512
#define GRP   32
#define CG    16
#define HW    4096          /* 64*64 */
#define MTOT  8192          /* B*HW */
#define EPSV  1e-6f
#define QSCALE 0.044194173824159216f   /* 1/sqrt(512) */

static __device__ __forceinline__ v16bf cat8(v8bf lo, v8bf hi) {
  return __builtin_shufflevector(lo, hi, 0,1,2,3,4,5,6,7,8,9,10,11,12,13,14,15);
}
static __device__ __forceinline__ v8f vzero8() {
  v8f z = {0.f,0.f,0.f,0.f,0.f,0.f,0.f,0.f};
  return z;
}
static __device__ __forceinline__ v8f wmma_bf16(v16bf a, v16bf b, v8f c) {
  // D = A(16x32 bf16) * B(32x16 bf16) + C(16x16 f32)
  return __builtin_amdgcn_wmma_f32_16x16x32_bf16(false, a, false, b, (short)0, c, false, false);
}

// ---- Tensor Data Mover: 2D bf16 tile, global -> LDS, with per-row LDS padding
// pad_interval_code 7 = 256 DWORDs (1024B = one 512-elem bf16 row)
// pad_amount_code   3 = 4 DWORDs (16B = 8 bf16 elems) -> LDS row stride 520 elems
static __device__ __forceinline__ void tdm_load_2d(const void* lds_ptr, const void* gptr,
                                                   unsigned tensor_d0, unsigned tensor_d1,
                                                   unsigned tile_d0, unsigned tile_d1,
                                                   unsigned stride_d0) {
  unsigned long long ga = (unsigned long long)(size_t)gptr;
  unsigned lds_off = (unsigned)(size_t)lds_ptr;   // low 32 bits of LDS aperture addr
  v4u g0;
  g0[0] = 1u;                                    // count=1 (user descriptor)
  g0[1] = lds_off;                               // lds_addr
  g0[2] = (unsigned)ga;                          // global_addr[31:0]
  g0[3] = (unsigned)(ga >> 32) | (2u << 30);     // global_addr[56:32] | type=2
  v8u g1;
  g1[0] = (1u << 16)       // data_size = 1 -> 2 bytes
        | (1u << 20)       // pad_enable
        | (7u << 22)       // pad_interval code (256 DWORDs)
        | (3u << 25);      // pad_amount code (4 DWORDs)
  g1[1] = (tensor_d0 & 0xFFFFu) << 16;                       // tensor_dim0[15:0]
  g1[2] = (tensor_d0 >> 16) | ((tensor_d1 & 0xFFFFu) << 16); // dim0 hi | dim1 lo
  g1[3] = (tensor_d1 >> 16) | (tile_d0 << 16);               // dim1 hi | tile_dim0
  g1[4] = tile_d1 & 0xFFFFu;                                 // tile_dim1 (tile_dim2=0)
  g1[5] = stride_d0;                                         // tensor_dim0_stride lo32
  g1[6] = 0u;                                                // stride hi | dim1_stride lo
  g1[7] = 0u;
  asm volatile("tensor_load_to_lds %0, %1" :: "s"(g0), "s"(g1) : "memory");
}

// ---------------------------------------------------------------- K1: GN stats
__global__ void gn_stats_kernel(const float* __restrict__ in, float* __restrict__ stats) {
  int bg = blockIdx.x;            // B*G = 64
  int b = bg >> 5, g = bg & 31;
  const float* base = in + (size_t)b * HW * C_DIM + g * CG;
  float s1 = 0.f, s2 = 0.f;
  for (int i = threadIdx.x; i < HW * CG; i += 256) {
    int sp = i >> 4, c = i & 15;
    float v = base[(size_t)sp * C_DIM + c];
    s1 += v; s2 += v * v;
  }
  __shared__ float r1[256], r2[256];
  r1[threadIdx.x] = s1; r2[threadIdx.x] = s2;
  __syncthreads();
  for (int st = 128; st > 0; st >>= 1) {
    if (threadIdx.x < st) {
      r1[threadIdx.x] += r1[threadIdx.x + st];
      r2[threadIdx.x] += r2[threadIdx.x + st];
    }
    __syncthreads();
  }
  if (threadIdx.x == 0) {
    float inv = 1.f / (float)(HW * CG);
    float mean = r1[0] * inv;
    float var  = r2[0] * inv - mean * mean;
    stats[bg * 2 + 0] = mean;
    stats[bg * 2 + 1] = rsqrtf(var + EPSV);
  }
}

// ------------------------------------------------------- K2: normalize -> bf16
__global__ void gn_apply_kernel(const float* __restrict__ in,
                                const float* __restrict__ gamma,
                                const float* __restrict__ beta,
                                const float* __restrict__ stats,
                                bf16* __restrict__ xb) {
  size_t i = (size_t)blockIdx.x * blockDim.x + threadIdx.x;
  size_t total  = (size_t)MTOT * C_DIM;          // 2^22
  size_t stride = (size_t)gridDim.x * blockDim.x;
  for (; i < total; i += stride) {
    int c = (int)(i & (C_DIM - 1));
    int b = (int)(i >> 21);                      // HW*C = 2^21 per batch
    int g = c >> 4;
    float mean = stats[(b * GRP + g) * 2 + 0];
    float rstd = stats[(b * GRP + g) * 2 + 1];
    float v = (in[i] - mean) * rstd * gamma[c] + beta[c];
    xb[i] = (bf16)v;
  }
}

// --------------------------------------- K2b: weights -> bf16, transposed (NxK)
__global__ void wconv_kernel(const float* __restrict__ wq, const float* __restrict__ wk,
                             const float* __restrict__ wv, const float* __restrict__ wo,
                             bf16* __restrict__ tq, bf16* __restrict__ tk,
                             bf16* __restrict__ tv, bf16* __restrict__ to_) {
  const float* src; bf16* dst;
  switch (blockIdx.z) {
    case 0:  src = wq; dst = tq;  break;
    case 1:  src = wk; dst = tk;  break;
    case 2:  src = wv; dst = tv;  break;
    default: src = wo; dst = to_; break;
  }
  int idx = blockIdx.x * 256 + threadIdx.x;      // grid.x = 1024 -> 262144
  int n = idx >> 9, c = idx & 511;
  dst[n * C_DIM + c] = (bf16)src[c * C_DIM + n]; // Wt[n][c] = W[c][n]
}

#define SWS 520  /* padded LDS row stride for 512-elem bf16 rows (TDM pad adds 16B/row) */

// ------------------------------------------------------------- K3: QKV GEMM
__global__ void qkv_gemm_kernel(const bf16* __restrict__ xb,
                                const bf16* __restrict__ tq, const bf16* __restrict__ tk,
                                const bf16* __restrict__ tv,
                                const float* __restrict__ bq, const float* __restrict__ bk,
                                const float* __restrict__ bv,
                                bf16* __restrict__ qo, bf16* __restrict__ ko,
                                bf16* __restrict__ vo) {
  const bf16* wt; const float* bias; bf16* out; float scale;
  switch (blockIdx.z) {
    case 0:  wt = tq; bias = bq; out = qo; scale = QSCALE; break;
    case 1:  wt = tk; bias = bk; out = ko; scale = 1.f;    break;
    default: wt = tv; bias = bv; out = vo; scale = 1.f;    break;
  }
  __shared__ bf16 sW[64 * SWS];                  // 64 output-channel rows of Wt
  int lane = threadIdx.x & 31, wave = threadIdx.x >> 5;
  int l16 = lane & 15, half = lane >> 4;
  int m0 = blockIdx.x * 128 + wave * 16;
  int n0 = blockIdx.y * 64;

  if (threadIdx.x < 32) {                        // TDM: stage Wt[n0:n0+64, :] in LDS
    tdm_load_2d(sW, wt + (size_t)n0 * C_DIM, C_DIM, C_DIM, C_DIM, 64, C_DIM);
    __builtin_amdgcn_s_wait_tensorcnt(0);
  }
  __syncthreads();

  v8f acc[4];
  for (int t = 0; t < 4; ++t) acc[t] = vzero8();

  const bf16* arow = xb + (size_t)(m0 + l16) * C_DIM;
  for (int kk = 0; kk < 16; ++kk) {
    int c0 = kk * 32;
    // A-frag (16x32): lane half selects K chunks {0..7,16..23} / {8..15,24..31}
    v16bf a = cat8(*(const v8bf*)(arow + c0 + half * 8),
                   *(const v8bf*)(arow + c0 + 16 + half * 8));
#pragma unroll
    for (int t = 0; t < 4; ++t) {
      // B-frag (32x16): lane = column; half selects K {0..7,16..23}/{8..15,24..31}
      const bf16* brow = &sW[(t * 16 + l16) * SWS + c0 + half * 8];
      v16bf b = cat8(*(const v8bf*)brow, *(const v8bf*)(brow + 16));
      acc[t] = wmma_bf16(a, b, acc[t]);
    }
  }
#pragma unroll
  for (int t = 0; t < 4; ++t) {
    int n = n0 + t * 16 + l16;
    float bb = bias[n];
#pragma unroll
    for (int vv = 0; vv < 8; ++vv) {
      int m = m0 + vv + half * 8;
      out[(size_t)m * C_DIM + n] = (bf16)((acc[t][vv] + bb) * scale);
    }
  }
}

// -------------------------------------------------------- K4: flash attention
#define SQS 520   /* LDS row strides (elements), 16B-aligned row bases */
#define SKS 520
#define SVS 40
#define SSS 40
#define SPS 40

__global__ void flash_attn_kernel(const bf16* __restrict__ q, const bf16* __restrict__ k,
                                  const bf16* __restrict__ v, bf16* __restrict__ attn) {
  __shared__ bf16  sQ [64 * SQS];
  __shared__ bf16  sK [32 * SKS];
  __shared__ bf16  sVt[512 * SVS];
  __shared__ float sS [64 * SSS];
  __shared__ bf16  sP [64 * SPS];
  __shared__ float sM[64], sL[64], sAlpha[64];

  int b = blockIdx.x >> 6, qb = blockIdx.x & 63;
  int qbase = qb * 64;
  int tid = threadIdx.x;
  int lane = tid & 31, wave = tid >> 5;
  int l16 = lane & 15, half = lane >> 4;
  const bf16* qp = q + ((size_t)b * HW + qbase) * C_DIM;
  const bf16* kp = k + (size_t)b * HW * C_DIM;
  const bf16* vp = v + (size_t)b * HW * C_DIM;

  // stage Q block (64 x 512) via TDM
  if (tid < 32)
    tdm_load_2d(sQ, qp, C_DIM, HW, C_DIM, 64, C_DIM);
  if (tid < 64) { sM[tid] = -__builtin_inff(); sL[tid] = 0.f; }

  v8f acc[4][4];
  for (int r = 0; r < 4; ++r)
    for (int c = 0; c < 4; ++c) acc[r][c] = vzero8();

  int wr = wave >> 1;            // score subtile row tile (0..3)
  int wc = (wave & 1) * 16;      // score subtile col base within 32
  int cw = wave * 64;            // this wave's V-column chunk

  for (int kt = 0; kt < 128; ++kt) {
    int kbase = kt * 32;
    __syncthreads();                                   // prior tile fully consumed
    if (tid < 32)                                      // TDM: K tile 32x512 -> LDS
      tdm_load_2d(sK, kp + (size_t)kbase * C_DIM, C_DIM, HW, C_DIM, 32, C_DIM);
    for (int j = 0; j < 64; ++j) {                     // V tile transposed -> 512x32
      int idx = tid + j * 256;
      int key = idx >> 9, cc = idx & 511;
      sVt[cc * SVS + key] = vp[(size_t)(kbase + key) * C_DIM + cc];
    }
    if (kt + 1 < 128) {                                // global_prefetch_b8 next tiles
      const char* nk = (const char*)(kp + (size_t)(kbase + 32) * C_DIM);
      const char* nv = (const char*)(vp + (size_t)(kbase + 32) * C_DIM);
      __builtin_prefetch(nk + tid * 128, 0, 0);
      __builtin_prefetch(nv + tid * 128, 0, 0);
    }
    if (tid < 32) __builtin_amdgcn_s_wait_tensorcnt(0); // Q (iter 0) + K tile done
    __syncthreads();

    // scores: each wave one 16x16 subtile, K = 512
    v8f s = vzero8();
    {
      const bf16* qrow = &sQ[(wr * 16 + l16) * SQS];
      const bf16* krow = &sK[(wc + l16) * SKS];
      for (int kk = 0; kk < 16; ++kk) {
        int c0 = kk * 32;
        v16bf a  = cat8(*(const v8bf*)(qrow + c0 + half * 8),
                        *(const v8bf*)(qrow + c0 + 16 + half * 8));
        v16bf bb = cat8(*(const v8bf*)(krow + c0 + half * 8),
                        *(const v8bf*)(krow + c0 + 16 + half * 8));
        s = wmma_bf16(a, bb, s);
      }
    }
#pragma unroll
    for (int vv = 0; vv < 8; ++vv)
      sS[(wr * 16 + vv + half * 8) * SSS + wc + l16] = s[vv];
    __syncthreads();

    // online softmax: 4 threads per row, 8 cols each
    {
      int row = tid >> 2, q4 = tid & 3;
      float* srow = &sS[row * SSS + q4 * 8];
      float mx = srow[0];
      for (int j = 1; j < 8; ++j) mx = fmaxf(mx, srow[j]);
      mx = fmaxf(mx, __shfl_xor(mx, 1, 4));
      mx = fmaxf(mx, __shfl_xor(mx, 2, 4));
      float mold = sM[row];
      float mnew = fmaxf(mold, mx);
      float sum = 0.f;
      bf16* prow = &sP[row * SPS + q4 * 8];
      for (int j = 0; j < 8; ++j) {
        float p = __expf(srow[j] - mnew);
        sum += p;
        prow[j] = (bf16)p;
      }
      sum += __shfl_xor(sum, 1, 4);
      sum += __shfl_xor(sum, 2, 4);
      if (q4 == 0) {
        float alpha = __expf(mold - mnew);
        sAlpha[row] = alpha;
        sL[row] = sL[row] * alpha + sum;
        sM[row] = mnew;
      }
    }
    __syncthreads();

    // rescale accumulators + P(64x32) @ V(32x64-chunk)
#pragma unroll
    for (int r = 0; r < 4; ++r) {
      float al[8];
#pragma unroll
      for (int vv = 0; vv < 8; ++vv) al[vv] = sAlpha[r * 16 + vv + half * 8];
#pragma unroll
      for (int cc = 0; cc < 4; ++cc)
#pragma unroll
        for (int vv = 0; vv < 8; ++vv) acc[r][cc][vv] *= al[vv];

      const bf16* prow = &sP[(r * 16 + l16) * SPS];
      v16bf aP = cat8(*(const v8bf*)(prow + half * 8),
                      *(const v8bf*)(prow + 16 + half * 8));
#pragma unroll
      for (int cc = 0; cc < 4; ++cc) {
        const bf16* vrow = &sVt[(cw + cc * 16 + l16) * SVS + half * 8];
        v16bf bV = cat8(*(const v8bf*)vrow, *(const v8bf*)(vrow + 16));
        acc[r][cc] = wmma_bf16(aP, bV, acc[r][cc]);
      }
    }
  }
  __syncthreads();

  // normalize by 1/l and store attn (bf16)
  bf16* op = attn + ((size_t)b * HW + qbase) * C_DIM;
#pragma unroll
  for (int r = 0; r < 4; ++r) {
    float rl[8];
#pragma unroll
    for (int vv = 0; vv < 8; ++vv) rl[vv] = 1.f / sL[r * 16 + vv + half * 8];
#pragma unroll
    for (int cc = 0; cc < 4; ++cc) {
      int col = cw + cc * 16 + l16;
#pragma unroll
      for (int vv = 0; vv < 8; ++vv) {
        int row = r * 16 + vv + half * 8;
        op[(size_t)row * C_DIM + col] = (bf16)(acc[r][cc][vv] * rl[vv]);
      }
    }
  }
}

// ---------------------------------------------- K5: out projection + residual
__global__ void out_gemm_kernel(const bf16* __restrict__ attn, const bf16* __restrict__ two,
                                const float* __restrict__ bo, const float* __restrict__ inputs,
                                float* __restrict__ out) {
  __shared__ bf16 sW[64 * SWS];
  int lane = threadIdx.x & 31, wave = threadIdx.x >> 5;
  int l16 = lane & 15, half = lane >> 4;
  int m0 = blockIdx.x * 128 + wave * 16;
  int n0 = blockIdx.y * 64;

  if (threadIdx.x < 32) {
    tdm_load_2d(sW, two + (size_t)n0 * C_DIM, C_DIM, C_DIM, C_DIM, 64, C_DIM);
    __builtin_amdgcn_s_wait_tensorcnt(0);
  }
  __syncthreads();

  v8f acc[4];
  for (int t = 0; t < 4; ++t) acc[t] = vzero8();

  const bf16* arow = attn + (size_t)(m0 + l16) * C_DIM;
  for (int kk = 0; kk < 16; ++kk) {
    int c0 = kk * 32;
    v16bf a = cat8(*(const v8bf*)(arow + c0 + half * 8),
                   *(const v8bf*)(arow + c0 + 16 + half * 8));
#pragma unroll
    for (int t = 0; t < 4; ++t) {
      const bf16* brow = &sW[(t * 16 + l16) * SWS + c0 + half * 8];
      v16bf b = cat8(*(const v8bf*)brow, *(const v8bf*)(brow + 16));
      acc[t] = wmma_bf16(a, b, acc[t]);
    }
  }
#pragma unroll
  for (int t = 0; t < 4; ++t) {
    int n = n0 + t * 16 + l16;
    float bb = bo[n];
#pragma unroll
    for (int vv = 0; vv < 8; ++vv) {
      size_t m = (size_t)(m0 + vv + half * 8);
      out[m * C_DIM + n] = acc[t][vv] + bb + inputs[m * C_DIM + n];
    }
  }
}

// ----------------------------------------------------------------- launcher
extern "C" void kernel_launch(void* const* d_in, const int* in_sizes, int n_in,
                              void* d_out, int out_size, void* d_ws, size_t ws_size,
                              hipStream_t stream) {
  const float* inputs = (const float*)d_in[0];
  const float* gamma  = (const float*)d_in[1];
  const float* beta   = (const float*)d_in[2];
  const float* wq = (const float*)d_in[3];
  const float* bq = (const float*)d_in[4];
  const float* wk = (const float*)d_in[5];
  const float* bk = (const float*)d_in[6];
  const float* wv = (const float*)d_in[7];
  const float* bv = (const float*)d_in[8];
  const float* wo = (const float*)d_in[9];
  const float* bo = (const float*)d_in[10];
  float* out = (float*)d_out;

  char* ws = (char*)d_ws;
  size_t off = 0;
  float* stats = (float*)ws;              off += 4096;
  bf16* xb  = (bf16*)(ws + off);          off += (size_t)MTOT * C_DIM * 2;   // 8 MB
  bf16* tq  = (bf16*)(ws + off);          off += (size_t)C_DIM * C_DIM * 2;  // 512 KB
  bf16* tk  = (bf16*)(ws + off);          off += (size_t)C_DIM * C_DIM * 2;
  bf16* tv  = (bf16*)(ws + off);          off += (size_t)C_DIM * C_DIM * 2;
  bf16* to_ = (bf16*)(ws + off);          off += (size_t)C_DIM * C_DIM * 2;
  bf16* qb_ = (bf16*)(ws + off);          off += (size_t)MTOT * C_DIM * 2;
  bf16* kb_ = (bf16*)(ws + off);          off += (size_t)MTOT * C_DIM * 2;
  bf16* vb_ = (bf16*)(ws + off);          off += (size_t)MTOT * C_DIM * 2;
  bf16* at_ = (bf16*)(ws + off);          off += (size_t)MTOT * C_DIM * 2;
  (void)in_sizes; (void)n_in; (void)out_size; (void)ws_size; (void)off;

  gn_stats_kernel<<<64, 256, 0, stream>>>(inputs, stats);
  gn_apply_kernel<<<4096, 256, 0, stream>>>(inputs, gamma, beta, stats, xb);
  wconv_kernel<<<dim3(1024, 1, 4), 256, 0, stream>>>(wq, wk, wv, wo, tq, tk, tv, to_);
  qkv_gemm_kernel<<<dim3(64, 8, 3), 256, 0, stream>>>(xb, tq, tk, tv, bq, bk, bv,
                                                      qb_, kb_, vb_);
  flash_attn_kernel<<<128, 256, 0, stream>>>(qb_, kb_, vb_, at_);
  out_gemm_kernel<<<dim3(64, 8), 256, 0, stream>>>(at_, to_, bo, inputs, out);
}
